// RelPosSelfAttention_86560770883670
// MI455X (gfx1250) — matrix operand order repaired and checked by reading, not compiled, not run on hardware
//
#include <hip/hip_runtime.h>

// ---------------------------------------------------------------------------
// RelPos self-attention for MI455X (gfx1250, wave32, WMMA f16->f32)
// Pipeline: cvt(f32->f16, W transposed) -> QKV GEMM -> fused attention -> out GEMM
// GEMMs stage W tiles with GLOBAL_LOAD_ASYNC_TO_LDS_B128 (ASYNCcnt) and
// compute 32x64 wave tiles with v_wmma_f32_16x16x32_f16.
// ---------------------------------------------------------------------------

typedef _Float16 f16;
typedef __attribute__((ext_vector_type(8)))  _Float16 v8h;
typedef __attribute__((ext_vector_type(16))) _Float16 v16h;
typedef __attribute__((ext_vector_type(8)))  float    v8f;

static constexpr int BATCH = 2048;
static constexpr int SEQ   = 64;
static constexpr int CH    = 1024;
static constexpr int NHEAD = 16;
static constexpr int DH    = 64;
static constexpr int MROWS = BATCH * SEQ;          // 131072

// ---- workspace layout (bytes) ----
static constexpr size_t OFF_XH   = 0;                                  // x f16 [M,1024]; later reused as attn-out f16
static constexpr size_t SZ_XH    = (size_t)MROWS * CH * 2;
static constexpr size_t OFF_WQ   = OFF_XH + SZ_XH;                     // W_qkv^T f16 [3072,1024]
static constexpr size_t SZ_WQ    = (size_t)3 * CH * CH * 2;
static constexpr size_t OFF_WO   = OFF_WQ + SZ_WQ;                     // W_out^T f16 [1024,1024]
static constexpr size_t SZ_WO    = (size_t)CH * CH * 2;
static constexpr size_t OFF_BIAS = OFF_WO + SZ_WO;                     // bias f32 [16,64,64]
static constexpr size_t SZ_BIAS  = (size_t)NHEAD * SEQ * SEQ * 4;
static constexpr size_t OFF_Q    = OFF_BIAS + SZ_BIAS;                 // q  f16 [B,H,L,dh]
static constexpr size_t SZ_HMAT  = (size_t)BATCH * NHEAD * SEQ * DH * 2;
static constexpr size_t OFF_K    = OFF_Q + SZ_HMAT;                    // k  f16 [B,H,L,dh]
static constexpr size_t OFF_VT   = OFF_K + SZ_HMAT;                    // v^T f16 [B,H,dh,L]

union HF16 { v16h v; v8h h[2]; };

__device__ __forceinline__ v16h make_frag(v8h lo, v8h hi) {
  HF16 u; u.h[0] = lo; u.h[1] = hi; return u.v;
}
__device__ __forceinline__ v8f vzero8() {
  v8f z;
#pragma unroll
  for (int i = 0; i < 8; ++i) z[i] = 0.0f;
  return z;
}
// A operand (16x32, row-major source): per-lane row fixed, K offsets {hi*8, 16+hi*8}
__device__ __forceinline__ v16h load_a(const f16* rowp, int kk, int hi) {
  return make_frag(*(const v8h*)(rowp + kk + hi * 8),
                   *(const v8h*)(rowp + kk + 16 + hi * 8));
}
// B operand (32x16, column stored as contiguous K row): K offsets {hi*8, 16+hi*8}
__device__ __forceinline__ v16h load_b(const f16* rowp, int kk, int hi) {
  return make_frag(*(const v8h*)(rowp + kk + hi * 8),
                   *(const v8h*)(rowp + kk + 16 + hi * 8));
}
#define WMMA_F16(a, b, c) \
  __builtin_amdgcn_wmma_f32_16x16x32_f16(false, (a), false, (b), (short)0, (c), false, false)

// async copy: per-lane 16B global -> LDS (VDST holds wave-relative LDS byte offset)
__device__ __forceinline__ void async_g2l_b128(unsigned lds_off, const f16* gptr) {
  asm volatile("global_load_async_to_lds_b128 %0, %1, off"
               :: "v"(lds_off), "v"((unsigned long long)(uintptr_t)gptr)
               : "memory");
}
__device__ __forceinline__ void wait_asynccnt0() {
  asm volatile("s_wait_asynccnt 0x0" ::: "memory");
}

// ---------------------------------------------------------------------------
// Stage 0: conversions
// ---------------------------------------------------------------------------
__global__ __launch_bounds__(256) void cvt_f32_to_f16x4(const float* __restrict__ src,
                                                        f16* __restrict__ dst, size_t n4) {
  size_t i = (size_t)blockIdx.x * blockDim.x + threadIdx.x;
  if (i >= n4) return;
  size_t b = i * 4;
  dst[b + 0] = (f16)src[b + 0];
  dst[b + 1] = (f16)src[b + 1];
  dst[b + 2] = (f16)src[b + 2];
  dst[b + 3] = (f16)src[b + 3];
}

// WT[n*rows + k] = (f16) W[k*cols + n]   (rows = K, cols = N)
__global__ __launch_bounds__(256) void cvt_transpose_f16(const float* __restrict__ W,
                                                         f16* __restrict__ WT,
                                                         int rows, int cols) {
  long long id = (long long)blockIdx.x * blockDim.x + threadIdx.x;
  if (id >= (long long)rows * cols) return;
  int k = (int)(id % rows);
  int n = (int)(id / rows);
  WT[(size_t)n * rows + k] = (f16)W[(size_t)k * cols + n];
}

// bias[h][m][n] from rel_bias[15,15,H] via 8x8 board coords
__global__ __launch_bounds__(256) void build_bias(const float* __restrict__ rel,
                                                  float* __restrict__ biasT) {
  int idx = blockIdx.x * 256 + threadIdx.x;            // 16*64*64 = 65536
  if (idx >= NHEAD * SEQ * SEQ) return;
  int h = idx >> 12;
  int m = (idx >> 6) & 63;
  int n = idx & 63;
  int ri = m >> 3, ci = m & 7;
  int rj = n >> 3, cj = n & 7;
  int dr = ri - rj + 7;
  int dc = ci - cj + 7;
  biasT[idx] = rel[(dr * 15 + dc) * NHEAD + h];
}

// ---------------------------------------------------------------------------
// Shared GEMM core: 256 thr = 8 waves; wave tile 32x64 (2x4 accums, 8 wmma/step).
// W tile (64 x 32 f16) double-buffered in LDS via async-to-LDS copies.
// k-loop kept rolled (unroll disabled) so accumulators keep one register
// placement and WMMA accumulates strictly in place (no rotation copies).
// lbs layout: [2][64][40] f16 (80B rows keep 16B-aligned v8h reads).
// ---------------------------------------------------------------------------
static constexpr int LROW  = 40;                       // padded row, f16 elems
static constexpr int LBUF  = 64 * LROW;                // one buffer, f16 elems

// compute helper: 8 WMMAs against one staged buffer
__device__ __forceinline__ void mma_step(v8f acc[2][4], const f16* arow0, const f16* arow1,
                                         int kk, const f16* bufp, int hi, int l15) {
  v16h a0 = load_a(arow0, kk, hi);
  v16h a1 = load_a(arow1, kk, hi);
#pragma unroll
  for (int t = 0; t < 4; ++t) {
    v16h bf = load_b(bufp + (t * 16 + l15) * LROW, 0, hi);
    acc[0][t] = WMMA_F16(a0, bf, acc[0][t]);
    acc[1][t] = WMMA_F16(a1, bf, acc[1][t]);
  }
}

// full K loop with async double-buffered staging
__device__ __forceinline__ void gemm_kloop(v8f acc[2][4],
                                           const f16* arow0, const f16* arow1,
                                           const f16* bsrc,           // this thread's stage src (k=0)
                                           f16* lbs,                  // &lbs[0]
                                           int brow, int bch,
                                           int hi, int l15) {
  f16* slot = lbs + brow * LROW + bch;                 // this thread's 16B stage slot
  unsigned lst0 = (unsigned)(uintptr_t)slot;           // buffer 0 LDS offset
  unsigned lst1 = lst0 + (unsigned)(LBUF * sizeof(f16));

  async_g2l_b128(lst0, bsrc);                          // prime buffer 0 (k-step 0)
  wait_asynccnt0();
  __syncthreads();

#pragma clang loop unroll(disable)
  for (int ks = 0; ks < 31; ++ks) {
    int nxt = ks + 1;
    async_g2l_b128((nxt & 1) ? lst1 : lst0, bsrc + nxt * 32);
    mma_step(acc, arow0, arow1, ks * 32, lbs + (ks & 1) * LBUF, hi, l15);
    wait_asynccnt0();
    __syncthreads();
  }
  mma_step(acc, arow0, arow1, 31 * 32, lbs + LBUF, hi, l15);  // ks=31 -> buffer 1
}

// ---------------------------------------------------------------------------
// Stage 1: QKV GEMM. A = xh [131072,1024], B = WqT [3072,1024].
// Grid (512, 48); block tile 256x64. Output scattered into q/k/vT.
// ---------------------------------------------------------------------------
__global__ __launch_bounds__(256) void qkv_gemm(const f16* __restrict__ xh,
                                                const f16* __restrict__ wqT,
                                                f16* __restrict__ qh,
                                                f16* __restrict__ kh,
                                                f16* __restrict__ vth) {
  __shared__ f16 lbs[2 * LBUF];
  int tid  = threadIdx.x;
  int wave = tid >> 5, lane = tid & 31;
  int l15  = lane & 15, hi = lane >> 4;
  int mbase = blockIdx.x * 256 + wave * 32;
  int nbase = blockIdx.y * 64;

  int brow = tid >> 2;                                 // 0..63
  int bch  = (tid & 3) * 8;                            // 0,8,16,24
  const f16* bsrc  = wqT + (size_t)(nbase + brow) * CH + bch;
  const f16* arow0 = xh + (size_t)(mbase + l15) * CH;
  const f16* arow1 = arow0 + (size_t)16 * CH;

  v8f acc[2][4];
#pragma unroll
  for (int u = 0; u < 2; ++u)
#pragma unroll
    for (int t = 0; t < 4; ++t) acc[u][t] = vzero8();

  gemm_kloop(acc, arow0, arow1, bsrc, lbs, brow, bch, hi, l15);

  // scatter stores: this block's 64 columns are exactly one (s, head) pair
  int s = nbase >> 10;                                 // 0=q 1=k 2=v
  int h = (nbase & 1023) >> 6;
#pragma unroll
  for (int u = 0; u < 2; ++u) {
#pragma unroll
    for (int t = 0; t < 4; ++t) {
      int d = t * 16 + l15;
#pragma unroll
      for (int r = 0; r < 8; ++r) {
        int gr = mbase + u * 16 + r + hi * 8;
        int b  = gr >> 6, l = gr & 63;
        f16 val = (f16)acc[u][t][r];
        size_t hb = (size_t)(b * NHEAD + h) * (SEQ * DH);
        if (s == 0)      qh [hb + (size_t)l * DH + d] = val;
        else if (s == 1) kh [hb + (size_t)l * DH + d] = val;
        else             vth[hb + (size_t)d * SEQ + l] = val;   // v stored transposed
      }
    }
  }
}

// ---------------------------------------------------------------------------
// Stage 2: fused attention. One block (4 waves) per (b,h).
// S = q k^T * scale + bias -> softmax -> O = P v  (P staged via LDS).
// ---------------------------------------------------------------------------
__global__ __launch_bounds__(128) void attn_kernel(const f16* __restrict__ qh,
                                                   const f16* __restrict__ kh,
                                                   const f16* __restrict__ vth,
                                                   const float* __restrict__ biasT,
                                                   f16* __restrict__ oh) {
  __shared__ f16 lp[64][72];                           // 144B rows: 16B-aligned v8h reads
  int bh = blockIdx.x;
  int b  = bh >> 4, h = bh & 15;
  const f16*   qb = qh  + (size_t)bh * (SEQ * DH);
  const f16*   kb = kh  + (size_t)bh * (SEQ * DH);
  const f16*   vb = vth + (size_t)bh * (SEQ * DH);
  const float* bb = biasT + (size_t)h * (SEQ * SEQ);

  int tid = threadIdx.x, wave = tid >> 5, lane = tid & 31;
  int l15 = lane & 15, hi = lane >> 4;
  int rb  = wave * 16;

  // ---- S = q k^T ----
  v8f sacc[4];
#pragma unroll
  for (int t = 0; t < 4; ++t) sacc[t] = vzero8();
  const f16* qrow = qb + (size_t)(rb + l15) * DH;
#pragma unroll
  for (int kk = 0; kk < DH; kk += 32) {
    v16h qf = load_a(qrow, kk, hi);
#pragma unroll
    for (int t = 0; t < 4; ++t) {
      const f16* kp = kb + (size_t)(t * 16 + l15) * DH;
      v16h kf = load_b(kp, kk, hi);
      sacc[t] = WMMA_F16(qf, kf, sacc[t]);
    }
  }

  // ---- scale + bias + softmax (one row per half-wave per accumulator vgpr) ----
  const float scale = 0.125f;                          // dh^-0.5
  float lsum[8];
#pragma unroll
  for (int r = 0; r < 8; ++r) {
    int m = rb + r + 8 * hi;
    float rowmax = -3.0e38f;
#pragma unroll
    for (int t = 0; t < 4; ++t) {
      float sv = sacc[t][r] * scale + bb[m * SEQ + t * 16 + l15];
      sacc[t][r] = sv;
      rowmax = fmaxf(rowmax, sv);
    }
#pragma unroll
    for (int msk = 1; msk < 16; msk <<= 1)
      rowmax = fmaxf(rowmax, __shfl_xor(rowmax, msk, 32));
    float rs = 0.0f;
#pragma unroll
    for (int t = 0; t < 4; ++t) {
      float p = __expf(sacc[t][r] - rowmax);
      sacc[t][r] = p;
      rs += p;
    }
#pragma unroll
    for (int msk = 1; msk < 16; msk <<= 1)
      rs += __shfl_xor(rs, msk, 32);
    lsum[r] = rs;
#pragma unroll
    for (int t = 0; t < 4; ++t)
      lp[m][t * 16 + l15] = (f16)sacc[t][r];           // P -> LDS (transpose staging)
  }
  __syncthreads();

  // ---- O = P v ----
  v8f oacc[4];
#pragma unroll
  for (int t = 0; t < 4; ++t) oacc[t] = vzero8();
  const f16* prow = &lp[rb + l15][0];
#pragma unroll
  for (int kk = 0; kk < SEQ; kk += 32) {
    v16h pf = load_a(prow, kk, hi);                    // ds_load_b128 fragments
#pragma unroll
    for (int t = 0; t < 4; ++t) {
      const f16* vp = vb + (size_t)(t * 16 + l15) * SEQ;  // v^T row d, contiguous j
      v16h vf = load_b(vp, kk, hi);
      oacc[t] = WMMA_F16(pf, vf, oacc[t]);
    }
  }

  // ---- normalize + store [B, L, H*dh] ----
#pragma unroll
  for (int t = 0; t < 4; ++t) {
    int d = t * 16 + l15;
#pragma unroll
    for (int r = 0; r < 8; ++r) {
      int m = rb + r + 8 * hi;
      oh[((size_t)(b * SEQ + m)) * CH + h * DH + d] = (f16)(oacc[t][r] / lsum[r]);
    }
  }
}

// ---------------------------------------------------------------------------
// Stage 3: output projection. A = oh [131072,1024], B = WoT [1024,1024].
// Grid (512, 16); fp32 output + bias.
// ---------------------------------------------------------------------------
__global__ __launch_bounds__(256) void out_gemm(const f16* __restrict__ ah,
                                                const f16* __restrict__ woT,
                                                const float* __restrict__ bout,
                                                float* __restrict__ out) {
  __shared__ f16 lbs[2 * LBUF];
  int tid  = threadIdx.x;
  int wave = tid >> 5, lane = tid & 31;
  int l15  = lane & 15, hi = lane >> 4;
  int mbase = blockIdx.x * 256 + wave * 32;
  int nbase = blockIdx.y * 64;

  int brow = tid >> 2;
  int bch  = (tid & 3) * 8;
  const f16* bsrc  = woT + (size_t)(nbase + brow) * CH + bch;
  const f16* arow0 = ah + (size_t)(mbase + l15) * CH;
  const f16* arow1 = arow0 + (size_t)16 * CH;

  v8f acc[2][4];
#pragma unroll
  for (int u = 0; u < 2; ++u)
#pragma unroll
    for (int t = 0; t < 4; ++t) acc[u][t] = vzero8();

  gemm_kloop(acc, arow0, arow1, bsrc, lbs, brow, bch, hi, l15);

#pragma unroll
  for (int t = 0; t < 4; ++t) {
    int gc = nbase + t * 16 + l15;
    float bo = bout[gc];
#pragma unroll
    for (int u = 0; u < 2; ++u) {
#pragma unroll
      for (int r = 0; r < 8; ++r) {
        int gr = mbase + u * 16 + r + hi * 8;
        out[(size_t)gr * CH + gc] = acc[u][t][r] + bo;
      }
    }
  }
}

// ---------------------------------------------------------------------------
extern "C" void kernel_launch(void* const* d_in, const int* in_sizes, int n_in,
                              void* d_out, int out_size, void* d_ws, size_t ws_size,
                              hipStream_t stream) {
  const float* x    = (const float*)d_in[0];   // [2048,64,1024]
  const float* Wqkv = (const float*)d_in[1];   // [1024,3072]
  const float* Wout = (const float*)d_in[2];   // [1024,1024]
  const float* bout = (const float*)d_in[3];   // [1024]
  const float* rel  = (const float*)d_in[4];   // [15,15,16]
  float* out = (float*)d_out;

  char* ws   = (char*)d_ws;
  f16*   xh    = (f16*)(ws + OFF_XH);
  f16*   wqT   = (f16*)(ws + OFF_WQ);
  f16*   woT   = (f16*)(ws + OFF_WO);
  float* biasT = (float*)(ws + OFF_BIAS);
  f16*   qh    = (f16*)(ws + OFF_Q);
  f16*   kh    = (f16*)(ws + OFF_K);
  f16*   vth   = (f16*)(ws + OFF_VT);
  f16*   oh    = xh;                            // reuse xh buffer (dead after qkv_gemm)

  // stage 0: conversions + bias table
  {
    size_t n4 = (size_t)MROWS * CH / 4;
    cvt_f32_to_f16x4<<<(unsigned)((n4 + 255) / 256), 256, 0, stream>>>(x, xh, n4);
    long long eq = (long long)CH * 3 * CH;
    cvt_transpose_f16<<<(unsigned)((eq + 255) / 256), 256, 0, stream>>>(Wqkv, wqT, CH, 3 * CH);
    long long eo = (long long)CH * CH;
    cvt_transpose_f16<<<(unsigned)((eo + 255) / 256), 256, 0, stream>>>(Wout, woT, CH, CH);
    build_bias<<<(NHEAD * SEQ * SEQ + 255) / 256, 256, 0, stream>>>(rel, biasT);
  }
  // stage 1: qkv projection (M=131072, N=3072, K=1024)
  qkv_gemm<<<dim3(MROWS / 256, (3 * CH) / 64), 256, 0, stream>>>(xh, wqT, qh, kh, vth);
  // stage 2: attention, one block per (batch, head)
  attn_kernel<<<BATCH * NHEAD, 128, 0, stream>>>(qh, kh, vth, biasT, oh);
  // stage 3: output projection (M=131072, N=1024, K=1024)
  out_gemm<<<dim3(MROWS / 256, CH / 64), 256, 0, stream>>>(oh, woT, bout, out);
}
